// DGATBlock_23914377904870
// MI455X (gfx1250) — matrix operand order repaired and checked by reading, not compiled, not run on hardware
//
#include <hip/hip_runtime.h>
#include <math.h>

#define NN 325
#define BB 4
#define TT 96
#define DD 128
#define HH 8
#define DHH 16
#define MF 64

typedef __attribute__((ext_vector_type(16))) __bf16 v16bf;
typedef __attribute__((ext_vector_type(8)))  float  v8f;
typedef __attribute__((ext_vector_type(2)))  float  v2f;

union FragBF { v16bf v; uint4 q[2]; unsigned short s[16]; };

// ---- bf16 conversion helpers -----------------------------------------------
__device__ __forceinline__ unsigned short f2bf_sw(float f) {
  unsigned u = __builtin_bit_cast(unsigned, f);
  u += 0x7FFFu + ((u >> 16) & 1u);           // round-to-nearest-even
  return (unsigned short)(u >> 16);
}
__device__ __forceinline__ unsigned pk2(float lo, float hi) {
#if defined(__gfx1250__)
  unsigned r;
  asm("v_cvt_pk_bf16_f32 %0, %1, %2" : "=v"(r) : "v"(lo), "v"(hi));
  return r;
#else
  return (unsigned)f2bf_sw(lo) | ((unsigned)f2bf_sw(hi) << 16);
#endif
}
__device__ __forceinline__ unsigned short f2bf(float f) {
#if defined(__gfx1250__)
  return (unsigned short)(pk2(f, 0.0f) & 0xffffu);
#else
  return f2bf_sw(f);
#endif
}
__device__ __forceinline__ float bflo(unsigned u) {
  return __builtin_bit_cast(float, u << 16);
}
__device__ __forceinline__ float bfhi(unsigned u) {
  return __builtin_bit_cast(float, u & 0xffff0000u);
}
// hardware cos: v_cos_f32 computes cos(2*pi*x) -> pre-scale by 1/(2*pi)
__device__ __forceinline__ float hwcos(float x) {
  return __builtin_amdgcn_cosf(x * 0.15915494309189535f);
}

// ---------------------------------------------------------------------------
// prep: bf16 weight copies + zero the cross-block accumulators (every call)
// ---------------------------------------------------------------------------
__global__ void prep_kernel(const float* __restrict__ Wq, const float* __restrict__ Wk,
                            const float* __restrict__ Wv, const float* __restrict__ Wo,
                            unsigned short* __restrict__ WB,    // [4][128*128] q,k,v,o
                            float* __restrict__ num_sum, float* __restrict__ den_sum) {
  int gid = blockIdx.x * blockDim.x + threadIdx.x;
  if (gid < NN * HH * MF * DHH) num_sum[gid] = 0.0f;
  if (gid < NN * HH * MF)       den_sum[gid] = 0.0f;
  if (gid < 4 * DD * DD) {
    int w = gid >> 14, idx = gid & 16383;
    const float* src = (w == 0) ? Wq : (w == 1) ? Wk : (w == 2) ? Wv : Wo;
    WB[gid] = f2bf(src[idx]);
  }
}

// ---------------------------------------------------------------------------
// pass1: k,v projection -> kp -> num_sum (N,H,64,16), den_sum (N,H,64)
// grid (N, 4=batch chunk), 256 threads (8 waves)
// ---------------------------------------------------------------------------
__global__ __launch_bounds__(256) void pass1_kernel(
    const float* __restrict__ z,
    const unsigned short* __restrict__ WkB, const unsigned short* __restrict__ WvB,
    const float* __restrict__ bk, const float* __restrict__ bv,
    const float* __restrict__ P,
    float* __restrict__ num_sum, float* __restrict__ den_sum) {

  __shared__ __align__(16) unsigned short zb[32][136];   // z tile, bf16
  __shared__ __align__(16) float          kt[32][132];   // k tile, fp32 (pre-phi)
  __shared__ __align__(16) unsigned short vt[32][136];   // v tile, bf16
  __shared__ __align__(16) unsigned short kpT[512][40];  // kp transposed [h*64+m][bt]

  const int n    = blockIdx.x;
  const int b    = blockIdx.y;            // 96 bt-rows per chunk == one batch b
  const int tid  = threadIdx.x;
  const int wave = __builtin_amdgcn_readfirstlane(tid >> 5);  // wave-uniform SGPR
  const int lane = tid & 31;
  const int lm   = lane & 15;
  const int hk   = lane >> 4;
  const float cf = 0.17677669529663689f;  // sqrt(2/64)

  v8f acc[4];
  #pragma unroll
  for (int j = 0; j < 4; ++j)
    #pragma unroll
    for (int i = 0; i < 8; ++i) acc[j][i] = 0.0f;
  float dacc0 = 0.0f, dacc1 = 0.0f;

  for (int it = 0; it < 3; ++it) {
    const int t0 = it * 32;
    __syncthreads();
    { // stage z (32 x 128 fp32 -> bf16 LDS), fully coalesced b128 loads
      int row = tid >> 3, cs = (tid & 7) * 16;
      const float* src = z + (((size_t)(b * NN + n)) * TT + t0 + row) * DD + cs;
      float4 f0 = ((const float4*)src)[0];
      float4 f1 = ((const float4*)src)[1];
      float4 f2 = ((const float4*)src)[2];
      float4 f3 = ((const float4*)src)[3];
      uint4 o;
      o.x = pk2(f0.x, f0.y); o.y = pk2(f0.z, f0.w);
      o.z = pk2(f1.x, f1.y); o.w = pk2(f1.z, f1.w);
      *(uint4*)&zb[row][cs] = o;
      o.x = pk2(f2.x, f2.y); o.y = pk2(f2.z, f2.w);
      o.z = pk2(f3.x, f3.y); o.w = pk2(f3.z, f3.w);
      *(uint4*)&zb[row][cs + 8] = o;
      if (it + 1 < 3) __builtin_prefetch(src + 32 * DD, 0, 1);  // global_prefetch_b8
    }
    __syncthreads();
    // K/V projection: 32 (tensor,rowtile,coltile) jobs, v_wmma_f32_16x16x32_bf16
    // two independent accumulator chains to hide XDL latency
    for (int jj = 0; jj < 4; ++jj) {
      int job = wave * 4 + jj;
      int tensor = job >> 4, rt = (job >> 3) & 1, ct = job & 7;
      const unsigned short* WBm = tensor ? WvB : WkB;
      v8f c0, c1;
      #pragma unroll
      for (int i = 0; i < 8; ++i) { c0[i] = 0.0f; c1[i] = 0.0f; }
      #pragma unroll
      for (int kc = 0; kc < 4; ++kc) {
        FragBF a, bf;
        const unsigned short* ar = &zb[rt * 16 + lm][kc * 32 + hk * 8];
        a.q[0] = *(const uint4*)ar;
        a.q[1] = *(const uint4*)(ar + 16);
        const unsigned short* br = WBm + (kc * 32) * DD + ct * 16 + lm;
        #pragma unroll
        for (int s = 0; s < 16; ++s) bf.s[s] = br[((s & 8) * 2 + hk * 8 + (s & 7)) * DD];
        if (kc & 1)
          c1 = __builtin_amdgcn_wmma_f32_16x16x32_bf16(false, a.v, false, bf.v,
                                                       (short)0, c1, false, false);
        else
          c0 = __builtin_amdgcn_wmma_f32_16x16x32_bf16(false, a.v, false, bf.v,
                                                       (short)0, c0, false, false);
      }
      v8f c = c0 + c1;
      float bias = (tensor ? bv : bk)[ct * 16 + lm];
      if (tensor == 0) {   // scalar branch: wave-uniform predicate
        #pragma unroll
        for (int i = 0; i < 8; ++i)
          kt[rt * 16 + hk * 8 + i][ct * 16 + lm] = 2.0f * (c[i] + bias); // /sqrt(tau)
      } else {
        #pragma unroll
        for (int i = 0; i < 8; ++i)
          vt[rt * 16 + hk * 8 + i][ct * 16 + lm] = f2bf(c[i] + bias);
      }
    }
    __syncthreads();
    // kp = cf*cos(k_h @ P^T): 64 jobs via v_wmma_f32_16x16x4_f32 (K=16 in 4 steps)
    for (int jj = 0; jj < 8; ++jj) {
      int job = wave * 8 + jj;
      int h = job >> 3, mt = (job >> 1) & 3, rt = job & 1;
      v8f c0, c1;
      #pragma unroll
      for (int i = 0; i < 8; ++i) { c0[i] = 0.0f; c1[i] = 0.0f; }
      #pragma unroll
      for (int kc = 0; kc < 4; ++kc) {
        v2f a, bf;
        a.x = kt[rt * 16 + lm][h * 16 + kc * 4 + hk * 2 + 0];
        a.y = kt[rt * 16 + lm][h * 16 + kc * 4 + hk * 2 + 1];
        bf.x = P[(mt * 16 + lm) * DHH + kc * 4 + hk * 2 + 0];  // B = P^T
        bf.y = P[(mt * 16 + lm) * DHH + kc * 4 + hk * 2 + 1];
        if (kc & 1)
          c1 = __builtin_amdgcn_wmma_f32_16x16x4_f32(false, a, false, bf,
                                                     (short)0, c1, false, false);
        else
          c0 = __builtin_amdgcn_wmma_f32_16x16x4_f32(false, a, false, bf,
                                                     (short)0, c0, false, false);
      }
      v8f c = c0 + c1;
      uint4 o;
      o.x = pk2(cf * hwcos(c[0]), cf * hwcos(c[1]));
      o.y = pk2(cf * hwcos(c[2]), cf * hwcos(c[3]));
      o.z = pk2(cf * hwcos(c[4]), cf * hwcos(c[5]));
      o.w = pk2(cf * hwcos(c[6]), cf * hwcos(c[7]));
      *(uint4*)&kpT[h * 64 + mt * 16 + lm][rt * 16 + hk * 8] = o;
    }
    __syncthreads();
    { // den partial sums over bt: two full kpT rows per thread, b128 reads
      const uint4* r0 = (const uint4*)&kpT[tid][0];
      const uint4* r1 = (const uint4*)&kpT[tid + 256][0];
      float s0 = 0.0f, s1 = 0.0f;
      #pragma unroll
      for (int qd = 0; qd < 4; ++qd) {
        uint4 a = r0[qd];
        s0 += bflo(a.x) + bfhi(a.x) + bflo(a.y) + bfhi(a.y)
            + bflo(a.z) + bfhi(a.z) + bflo(a.w) + bfhi(a.w);
        uint4 bq4 = r1[qd];
        s1 += bflo(bq4.x) + bfhi(bq4.x) + bflo(bq4.y) + bfhi(bq4.y)
            + bflo(bq4.z) + bfhi(bq4.z) + bflo(bq4.w) + bfhi(bq4.w);
      }
      dacc0 += s0; dacc1 += s1;
    }
    // num_sum accumulation: (64x16) += kp^T(16x32) @ v(32x16) per (h, mtile)
    for (int jj = 0; jj < 4; ++jj) {
      int job = wave * 4 + jj;
      int h = job >> 2, mt = job & 3;
      FragBF a, bf;
      const unsigned short* ar = &kpT[h * 64 + mt * 16 + lm][0];
      a.q[0] = *(const uint4*)(ar + hk * 8);
      a.q[1] = *(const uint4*)(ar + 16 + hk * 8);
      const unsigned short* br = &vt[0][h * 16 + lm];
      #pragma unroll
      for (int s = 0; s < 16; ++s) bf.s[s] = br[((s & 8) * 2 + hk * 8 + (s & 7)) * 136];
      acc[jj] = __builtin_amdgcn_wmma_f32_16x16x32_bf16(false, a.v, false, bf.v,
                                                        (short)0, acc[jj], false, false);
    }
  }
  // flush block-local accumulators
  for (int jj = 0; jj < 4; ++jj) {
    int job = wave * 4 + jj;
    int h = job >> 2, mt = job & 3;
    #pragma unroll
    for (int i = 0; i < 8; ++i) {
      int m = mt * 16 + hk * 8 + i;
      atomicAdd(&num_sum[(((size_t)n * HH + h) * MF + m) * DHH + lm], acc[jj][i]);
    }
  }
  atomicAdd(&den_sum[(size_t)n * HH * MF + tid],       dacc0);
  atomicAdd(&den_sum[(size_t)n * HH * MF + tid + 256], dacc1);
}

// ---------------------------------------------------------------------------
// pass2: q proj -> qp -> numerator/denominator -> z_out -> out = z_out@Wo+bo
// grid (N, 24 bt-tiles), 128 threads (4 waves)
// ---------------------------------------------------------------------------
__global__ __launch_bounds__(128) void pass2_kernel(
    const float* __restrict__ z,
    const unsigned short* __restrict__ WqB, const float* __restrict__ bq,
    const float* __restrict__ P,
    const float* __restrict__ num_sum, const float* __restrict__ den_sum,
    const unsigned short* __restrict__ WoB, const float* __restrict__ bo,
    float* __restrict__ out) {

  __shared__ __align__(16) unsigned short zb[16][136];
  __shared__ __align__(16) float          qt[16][132];
  __shared__ __align__(16) unsigned short qp[16][520];   // [bt][h*64+m] bf16
  __shared__ __align__(16) float          dn[16][8];
  __shared__ __align__(16) unsigned short zo[16][136];   // z_out bf16

  const int n   = blockIdx.x;
  const int bt0 = blockIdx.y * 16;
  const int b   = bt0 / TT, t0v = bt0 % TT;
  const int tid = threadIdx.x;
  const int wave = __builtin_amdgcn_readfirstlane(tid >> 5);
  const int lane = tid & 31;
  const int lm = lane & 15, hk = lane >> 4;
  const float cf = 0.17677669529663689f;

  { // stage z (16 x 128)
    int row = tid >> 3, cs = (tid & 7) * 16;
    const float* src = z + (((size_t)(b * NN + n)) * TT + t0v + row) * DD + cs;
    float4 f0 = ((const float4*)src)[0];
    float4 f1 = ((const float4*)src)[1];
    float4 f2 = ((const float4*)src)[2];
    float4 f3 = ((const float4*)src)[3];
    uint4 o;
    o.x = pk2(f0.x, f0.y); o.y = pk2(f0.z, f0.w);
    o.z = pk2(f1.x, f1.y); o.w = pk2(f1.z, f1.w);
    *(uint4*)&zb[row][cs] = o;
    o.x = pk2(f2.x, f2.y); o.y = pk2(f2.z, f2.w);
    o.z = pk2(f3.x, f3.y); o.w = pk2(f3.z, f3.w);
    *(uint4*)&zb[row][cs + 8] = o;
  }
  __syncthreads();
  // q projection
  for (int jc = 0; jc < 2; ++jc) {
    int ct = wave * 2 + jc;
    v8f c0, c1;
    #pragma unroll
    for (int i = 0; i < 8; ++i) { c0[i] = 0.0f; c1[i] = 0.0f; }
    #pragma unroll
    for (int kc = 0; kc < 4; ++kc) {
      FragBF a, bf;
      const unsigned short* ar = &zb[lm][kc * 32 + hk * 8];
      a.q[0] = *(const uint4*)ar;
      a.q[1] = *(const uint4*)(ar + 16);
      const unsigned short* br = WqB + (kc * 32) * DD + ct * 16 + lm;
      #pragma unroll
      for (int s = 0; s < 16; ++s) bf.s[s] = br[((s & 8) * 2 + hk * 8 + (s & 7)) * DD];
      if (kc & 1)
        c1 = __builtin_amdgcn_wmma_f32_16x16x32_bf16(false, a.v, false, bf.v,
                                                     (short)0, c1, false, false);
      else
        c0 = __builtin_amdgcn_wmma_f32_16x16x32_bf16(false, a.v, false, bf.v,
                                                     (short)0, c0, false, false);
    }
    v8f c = c0 + c1;
    float bias = bq[ct * 16 + lm];
    #pragma unroll
    for (int i = 0; i < 8; ++i)
      qt[hk * 8 + i][ct * 16 + lm] = 2.0f * (c[i] + bias);
  }
  __syncthreads();
  // qp = cf*cos(q_h @ P^T)
  for (int jj = 0; jj < 8; ++jj) {
    int job = wave * 8 + jj;
    int h = job >> 2, mt = job & 3;
    v8f c0, c1;
    #pragma unroll
    for (int i = 0; i < 8; ++i) { c0[i] = 0.0f; c1[i] = 0.0f; }
    #pragma unroll
    for (int kc = 0; kc < 4; ++kc) {
      v2f a, bf;
      a.x = qt[lm][h * 16 + kc * 4 + hk * 2 + 0];
      a.y = qt[lm][h * 16 + kc * 4 + hk * 2 + 1];
      bf.x = P[(mt * 16 + lm) * DHH + kc * 4 + hk * 2 + 0];
      bf.y = P[(mt * 16 + lm) * DHH + kc * 4 + hk * 2 + 1];
      if (kc & 1)
        c1 = __builtin_amdgcn_wmma_f32_16x16x4_f32(false, a, false, bf,
                                                   (short)0, c1, false, false);
      else
        c0 = __builtin_amdgcn_wmma_f32_16x16x4_f32(false, a, false, bf,
                                                   (short)0, c0, false, false);
    }
    v8f c = c0 + c1;
    #pragma unroll
    for (int i = 0; i < 8; ++i)
      qp[hk * 8 + i][h * 64 + mt * 16 + lm] = f2bf(cf * hwcos(c[i]));
  }
  __syncthreads();
  { // denominator: one (bt,h) per thread, vectorized b128 reads
    int btr = tid >> 3, h = tid & 7;
    const float4* ds4 = (const float4*)(den_sum + (size_t)n * HH * MF + h * MF);
    const uint4* qr = (const uint4*)&qp[btr][h * 64];
    float s = 0.0f;
    #pragma unroll
    for (int qd = 0; qd < 8; ++qd) {
      uint4 a = qr[qd];
      float4 d0 = ds4[qd * 2], d1 = ds4[qd * 2 + 1];
      s += bflo(a.x) * d0.x + bfhi(a.x) * d0.y + bflo(a.y) * d0.z + bfhi(a.y) * d0.w
         + bflo(a.z) * d1.x + bfhi(a.z) * d1.y + bflo(a.w) * d1.z + bfhi(a.w) * d1.w;
    }
    dn[btr][h] = s;
  }
  __syncthreads();
  // numerator: (16x16) = qp_h(16x64) @ num_sum_h(64x16); then divide
  for (int jh = 0; jh < 2; ++jh) {
    int h = wave * 2 + jh;
    v8f c0, c1;
    #pragma unroll
    for (int i = 0; i < 8; ++i) { c0[i] = 0.0f; c1[i] = 0.0f; }
    #pragma unroll
    for (int kc = 0; kc < 2; ++kc) {
      FragBF a, bf;
      const unsigned short* ar = &qp[lm][h * 64 + kc * 32 + hk * 8];
      a.q[0] = *(const uint4*)ar;
      a.q[1] = *(const uint4*)(ar + 16);
      const float* nr = num_sum + (((size_t)n * HH + h) * MF + kc * 32) * DHH + lm;
      #pragma unroll
      for (int s = 0; s < 16; ++s) bf.s[s] = f2bf(nr[((s & 8) * 2 + hk * 8 + (s & 7)) * DHH]);
      if (kc & 1)
        c1 = __builtin_amdgcn_wmma_f32_16x16x32_bf16(false, a.v, false, bf.v,
                                                     (short)0, c1, false, false);
      else
        c0 = __builtin_amdgcn_wmma_f32_16x16x32_bf16(false, a.v, false, bf.v,
                                                     (short)0, c0, false, false);
    }
    v8f c = c0 + c1;
    #pragma unroll
    for (int i = 0; i < 8; ++i) {
      float dv = dn[hk * 8 + i][h] + 1e-8f;
      zo[hk * 8 + i][h * 16 + lm] = f2bf(c[i] / dv);
    }
  }
  __syncthreads();
  // final: out tile = zo(16x128) @ Wo + bo
  for (int jc = 0; jc < 2; ++jc) {
    int ct = wave * 2 + jc;
    v8f c0, c1;
    #pragma unroll
    for (int i = 0; i < 8; ++i) { c0[i] = 0.0f; c1[i] = 0.0f; }
    #pragma unroll
    for (int kc = 0; kc < 4; ++kc) {
      FragBF a, bf;
      const unsigned short* ar = &zo[lm][kc * 32 + hk * 8];
      a.q[0] = *(const uint4*)ar;
      a.q[1] = *(const uint4*)(ar + 16);
      const unsigned short* br = WoB + (kc * 32) * DD + ct * 16 + lm;
      #pragma unroll
      for (int s = 0; s < 16; ++s) bf.s[s] = br[((s & 8) * 2 + hk * 8 + (s & 7)) * DD];
      if (kc & 1)
        c1 = __builtin_amdgcn_wmma_f32_16x16x32_bf16(false, a.v, false, bf.v,
                                                     (short)0, c1, false, false);
      else
        c0 = __builtin_amdgcn_wmma_f32_16x16x32_bf16(false, a.v, false, bf.v,
                                                     (short)0, c0, false, false);
    }
    v8f c = c0 + c1;
    float bias = bo[ct * 16 + lm];
    #pragma unroll
    for (int i = 0; i < 8; ++i)
      out[(((size_t)(b * NN + n)) * TT + t0v + hk * 8 + i) * DD + ct * 16 + lm] =
          c[i] + bias;
  }
}

// ---------------------------------------------------------------------------
extern "C" void kernel_launch(void* const* d_in, const int* in_sizes, int n_in,
                              void* d_out, int out_size, void* d_ws, size_t ws_size,
                              hipStream_t stream) {
  (void)in_sizes; (void)n_in; (void)out_size; (void)ws_size;
  const float* z  = (const float*)d_in[0];
  // d_in[1]=edge_index (unused), d_in[2]=M_theta (statically skipped: H != N)
  const float* P  = (const float*)d_in[3];
  const float* Wq = (const float*)d_in[4];  const float* bq = (const float*)d_in[5];
  const float* Wk = (const float*)d_in[6];  const float* bk = (const float*)d_in[7];
  const float* Wv = (const float*)d_in[8];  const float* bv = (const float*)d_in[9];
  const float* Wo = (const float*)d_in[10]; const float* bo = (const float*)d_in[11];
  float* out = (float*)d_out;

  char* ws = (char*)d_ws;
  unsigned short* WB = (unsigned short*)ws;              // 4 * 128*128 bf16 = 128 KB
  unsigned short* WqB = WB;
  unsigned short* WkB = WB + 1 * DD * DD;
  unsigned short* WvB = WB + 2 * DD * DD;
  unsigned short* WoB = WB + 3 * DD * DD;
  float* num_sum = (float*)(ws + 131072);                           // 10.65 MB
  float* den_sum = (float*)(ws + 131072 + (size_t)NN*HH*MF*DHH*4);  // 0.67 MB

  const int prep_total = NN * HH * MF * DHH;             // covers all prep work
  prep_kernel<<<(prep_total + 255) / 256, 256, 0, stream>>>(
      Wq, Wk, Wv, Wo, WB, num_sum, den_sum);

  pass1_kernel<<<dim3(NN, BB), 256, 0, stream>>>(
      z, WkB, WvB, bk, bv, P, num_sum, den_sum);

  pass2_kernel<<<dim3(NN, (BB * TT) / 16), 128, 0, stream>>>(
      z, WqB, bq, P, num_sum, den_sum, WoB, bo, out);
}
